// Transformer_89172110999925
// MI455X (gfx1250) — matrix operand order repaired and checked by reading, not compiled, not run on hardware
//
#include <hip/hip_runtime.h>
#include <math.h>

#define D_MODEL 768
#define SEQ     1024
#define NBATCH  32
#define NLAYER  6
#define NCLS    6
#define FFDIM   1536
#define LN_EPS  1e-5f

typedef __attribute__((ext_vector_type(16))) __bf16 v16bf;
typedef __attribute__((ext_vector_type(8)))  __bf16 v8bf;
typedef __attribute__((ext_vector_type(8)))  float  v8f;

// ---------------------------------------------------------------- embedding
__global__ void embed_kernel(const int* __restrict__ tokens,
                             const float* __restrict__ emb,
                             const float* __restrict__ pos,
                             float* __restrict__ x) {
  const int row = blockIdx.x;          // b*SEQ + s
  const int s   = row & (SEQ - 1);
  const int tok = tokens[row];
  const float* er = emb + (size_t)tok * D_MODEL;
  const float* pr = pos + (size_t)s * D_MODEL;
  float* xr = x + (size_t)row * D_MODEL;
  for (int d = threadIdx.x; d < D_MODEL; d += blockDim.x)
    xr[d] = er[d] + pr[d];
}

// ---------------------------------------------------------------- layernorm
__global__ void ln_kernel(const float* __restrict__ in,
                          const float* __restrict__ w,
                          const float* __restrict__ b,
                          float* __restrict__ out) {
  __shared__ float rs[256];
  __shared__ float rss[256];
  const int row = blockIdx.x;
  const int t = threadIdx.x;
  const float* xr = in + (size_t)row * D_MODEL;
  float vals[3];
  float s = 0.f, ss = 0.f;
#pragma unroll
  for (int i = 0; i < 3; ++i) {
    const int d = t + i * 256;
    vals[i] = xr[d];
    s += vals[i];
    ss += vals[i] * vals[i];
  }
  rs[t] = s; rss[t] = ss;
  __syncthreads();
  for (int o = 128; o > 0; o >>= 1) {
    if (t < o) { rs[t] += rs[t + o]; rss[t] += rss[t + o]; }
    __syncthreads();
  }
  const float mean = rs[0] * (1.0f / D_MODEL);
  const float var  = rss[0] * (1.0f / D_MODEL) - mean * mean;
  const float inv  = rsqrtf(var + LN_EPS);
  float* orow = out + (size_t)row * D_MODEL;
#pragma unroll
  for (int i = 0; i < 3; ++i) {
    const int d = t + i * 256;
    orow[d] = (vals[i] - mean) * inv * w[d] + b[d];
  }
}

// ---------------------------------------------------------------- WMMA GEMM
// C[z] = alpha * (A[z] @ B[z]) (+bias) (relu?) (+residual)
// flags: bit0 = relu, bit1 = B stored [N][K] (transposed)
#define BM 128
#define BN 128
#define BK 32
#define LDA_P 40     // bf16 elems per A row (k-major, padded)
#define LDB_P 40     // bf16 elems per B row (n-major, padded)

__global__ __launch_bounds__(256)
void gemm_wmma_kernel(const float* __restrict__ A,
                      const float* __restrict__ B,
                      const float* __restrict__ bias,
                      const float* __restrict__ residual,
                      float* __restrict__ C,
                      int M, int N, int K,
                      long long sA, long long sB, long long sC,
                      float alpha, int flags) {
  __shared__ __bf16 ldsA[2][BM * LDA_P];   // [m][k]
  __shared__ __bf16 ldsB[2][BN * LDB_P];   // [n][k]

  const int t    = threadIdx.x;
  const int lane = t & 31;
  const int wv   = t >> 5;
  const int half = lane >> 4;
  const int ln16 = lane & 15;
  const int wm   = (wv >> 2) * 64;   // wave M offset inside block tile
  const int wn   = (wv & 3) * 32;    // wave N offset inside block tile

  const int bm = blockIdx.y * BM;
  const int bn = blockIdx.x * BN;
  const int z  = blockIdx.z;
  const float* Ab = A + (size_t)z * (size_t)sA;
  const float* Bb = B + (size_t)z * (size_t)sB;
  float*       Cb = C + (size_t)z * (size_t)sC;
  const float* Rb = residual ? residual + (size_t)z * (size_t)sC : nullptr;

  const bool relu   = (flags & 1) != 0;
  const bool transB = (flags & 2) != 0;

  // staging coordinates: 256 threads cover 128 rows x {k 0..15, 16..31}
  const int am  = t >> 1;            // A row 0..127
  const int akg = (t & 1) << 4;      // A k-group 0 / 16
  const int bnl = t & 127;           // B local n 0..127
  const int bkg = (t >> 7) << 4;     // B k-group 0 / 16

  v8f acc[4][2];
  const v8f zero8 = {0.f, 0.f, 0.f, 0.f, 0.f, 0.f, 0.f, 0.f};
#pragma unroll
  for (int mf = 0; mf < 4; ++mf)
#pragma unroll
    for (int nf = 0; nf < 2; ++nf) acc[mf][nf] = zero8;

  float ra[16], rb[16];

  auto loadTiles = [&](int kb) {
    const float4* pa = (const float4*)(Ab + (size_t)(bm + am) * K + kb + akg);
#pragma unroll
    for (int i = 0; i < 4; ++i) {
      const float4 f = pa[i];
      ra[4 * i + 0] = f.x; ra[4 * i + 1] = f.y;
      ra[4 * i + 2] = f.z; ra[4 * i + 3] = f.w;
    }
    if (transB) {
      const float4* pb = (const float4*)(Bb + (size_t)(bn + bnl) * K + kb + bkg);
#pragma unroll
      for (int i = 0; i < 4; ++i) {
        const float4 f = pb[i];
        rb[4 * i + 0] = f.x; rb[4 * i + 1] = f.y;
        rb[4 * i + 2] = f.z; rb[4 * i + 3] = f.w;
      }
    } else {
#pragma unroll
      for (int e = 0; e < 16; ++e)
        rb[e] = Bb[(size_t)(kb + bkg + e) * N + bn + bnl];
    }
  };

  auto storeTiles = [&](int buf) {
    v8bf lo, hi;
#pragma unroll
    for (int i = 0; i < 8; ++i) { lo[i] = (__bf16)ra[i]; hi[i] = (__bf16)ra[8 + i]; }
    *(v8bf*)&ldsA[buf][am * LDA_P + akg]     = lo;
    *(v8bf*)&ldsA[buf][am * LDA_P + akg + 8] = hi;
#pragma unroll
    for (int i = 0; i < 8; ++i) { lo[i] = (__bf16)rb[i]; hi[i] = (__bf16)rb[8 + i]; }
    *(v8bf*)&ldsB[buf][bnl * LDB_P + bkg]     = lo;
    *(v8bf*)&ldsB[buf][bnl * LDB_P + bkg + 8] = hi;
  };

  const int kTiles = K / BK;
  loadTiles(0);
  storeTiles(0);
  __syncthreads();

  int cur = 0;
  for (int kt = 0; kt < kTiles; ++kt) {
    const bool hasNext = (kt + 1) < kTiles;
    if (hasNext) {
      loadTiles((kt + 1) * BK);
      if (kt + 2 < kTiles)   // gfx1250 global_prefetch_b8 for tile after next
        __builtin_prefetch(Ab + (size_t)(bm + am) * K + (size_t)(kt + 2) * BK + akg, 0, 1);
    }

    // A fragments: two contiguous 8-elem runs per lane (k = half*8+e, 16+half*8+e)
    v16bf afrag[4];
#pragma unroll
    for (int mf = 0; mf < 4; ++mf) {
      const int base = (wm + mf * 16 + ln16) * LDA_P + half * 8;
      const v8bf a0 = *(const v8bf*)&ldsA[cur][base];
      const v8bf a1 = *(const v8bf*)&ldsA[cur][base + 16];
      afrag[mf] = __builtin_shufflevector(a0, a1, 0, 1, 2, 3, 4, 5, 6, 7,
                                          8, 9, 10, 11, 12, 13, 14, 15);
    }
    // B fragments: two contiguous 8-elem runs per lane (k = half*8+e, 16+half*8+e)
    v16bf bfrag[2];
#pragma unroll
    for (int nf = 0; nf < 2; ++nf) {
      const int base = (wn + nf * 16 + ln16) * LDB_P + half * 8;
      const v8bf b0 = *(const v8bf*)&ldsB[cur][base];
      const v8bf b1 = *(const v8bf*)&ldsB[cur][base + 16];
      bfrag[nf] = __builtin_shufflevector(b0, b1, 0, 1, 2, 3, 4, 5, 6, 7,
                                          8, 9, 10, 11, 12, 13, 14, 15);
    }

#pragma unroll
    for (int mf = 0; mf < 4; ++mf)
#pragma unroll
      for (int nf = 0; nf < 2; ++nf)
        acc[mf][nf] = __builtin_amdgcn_wmma_f32_16x16x32_bf16(
            false, afrag[mf], false, bfrag[nf], (short)0, acc[mf][nf], false, false);

    if (hasNext) {
      storeTiles(cur ^ 1);   // other buffer: no barrier needed before store
      __syncthreads();       // visibility of next tile before next iteration
      cur ^= 1;
    }
  }

  // epilogue: C/D layout — VGPR r: row = r + 8*half, col = lane&15
#pragma unroll
  for (int mf = 0; mf < 4; ++mf) {
#pragma unroll
    for (int nf = 0; nf < 2; ++nf) {
      const int col = bn + wn + nf * 16 + ln16;
#pragma unroll
      for (int r = 0; r < 8; ++r) {
        const int row = bm + wm + mf * 16 + r + half * 8;
        float val = acc[mf][nf][r] * alpha;
        if (bias) val += bias[col];
        if (relu) val = fmaxf(val, 0.f);
        const size_t off = (size_t)row * N + col;
        if (Rb) val += Rb[off];
        Cb[off] = val;
      }
    }
  }
}

// ------------------------------------------------- masked softmax over keys
__global__ void softmax_mask_kernel(float* __restrict__ scores,
                                    const int* __restrict__ tokens) {
  __shared__ float red[256];
  const int b = blockIdx.x >> 10;            // row = b*SEQ + i
  float* row = scores + (size_t)blockIdx.x * SEQ;
  const int* tk = tokens + b * SEQ;
  const int t = threadIdx.x;
  const float NEG = -__builtin_inff();
  float vals[4];
  float mx = NEG;
#pragma unroll
  for (int i = 0; i < 4; ++i) {
    const int j = t + i * 256;
    const float v = (tk[j] == 0) ? NEG : row[j];
    vals[i] = v;
    mx = fmaxf(mx, v);
  }
  red[t] = mx; __syncthreads();
  for (int o = 128; o > 0; o >>= 1) {
    if (t < o) red[t] = fmaxf(red[t], red[t + o]);
    __syncthreads();
  }
  mx = red[0]; __syncthreads();
  float s = 0.f;
#pragma unroll
  for (int i = 0; i < 4; ++i) {
    const float e = (vals[i] == NEG) ? 0.f : __expf(vals[i] - mx);
    vals[i] = e; s += e;
  }
  red[t] = s; __syncthreads();
  for (int o = 128; o > 0; o >>= 1) {
    if (t < o) red[t] += red[t + o];
    __syncthreads();
  }
  const float inv = 1.f / fmaxf(red[0], 1e-20f);
#pragma unroll
  for (int i = 0; i < 4; ++i) row[t + i * 256] = vals[i] * inv;
}

// ---------------------------------------------------------- masked mean pool
__global__ void pool_kernel(const float* __restrict__ xn,
                            const int* __restrict__ tokens,
                            float* __restrict__ pooled) {
  __shared__ float red[256];
  const int b = blockIdx.x;
  const int t = threadIdx.x;
  const int* tk = tokens + b * SEQ;
  float cnt = 0.f;
  for (int s = t; s < SEQ; s += 256) cnt += (tk[s] != 0) ? 1.f : 0.f;
  red[t] = cnt; __syncthreads();
  for (int o = 128; o > 0; o >>= 1) {
    if (t < o) red[t] += red[t + o];
    __syncthreads();
  }
  const float denom = 1.f / fmaxf(red[0], 1.f);
  const float* xb = xn + (size_t)b * SEQ * D_MODEL;
  for (int d = t; d < D_MODEL; d += 256) {
    float acc = 0.f;
    for (int s = 0; s < SEQ; ++s)
      if (tk[s] != 0) acc += xb[(size_t)s * D_MODEL + d];
    pooled[b * D_MODEL + d] = acc * denom;
  }
}

// ---------------------------------------------------------------- classifier
__global__ void cls_kernel(const float* __restrict__ pooled,
                           const float* __restrict__ wc,
                           const float* __restrict__ bc,
                           float* __restrict__ out) {
  const int i = blockIdx.x * blockDim.x + threadIdx.x;
  if (i >= NBATCH * NCLS) return;
  const int b = i / NCLS, c = i % NCLS;
  float acc = bc[c];
  const float* pr = pooled + b * D_MODEL;
  for (int d = 0; d < D_MODEL; ++d) acc += pr[d] * wc[d * NCLS + c];
  out[i] = acc;
}

// ------------------------------------------------------------------- driver
extern "C" void kernel_launch(void* const* d_in, const int* in_sizes, int n_in,
                              void* d_out, int out_size, void* d_ws, size_t ws_size,
                              hipStream_t stream) {
  const int*   tokens = (const int*)  d_in[0];
  const float* emb    = (const float*)d_in[1];
  const float* pos    = (const float*)d_in[2];
  const float* ln1_w  = (const float*)d_in[3];
  const float* ln1_b  = (const float*)d_in[4];
  const float* ln2_w  = (const float*)d_in[5];
  const float* ln2_b  = (const float*)d_in[6];
  const float* wq     = (const float*)d_in[7];
  const float* bq     = (const float*)d_in[8];
  const float* wk     = (const float*)d_in[9];
  const float* bk     = (const float*)d_in[10];
  const float* wvv    = (const float*)d_in[11];
  const float* bv     = (const float*)d_in[12];
  const float* w1     = (const float*)d_in[13];
  const float* b1     = (const float*)d_in[14];
  const float* w2     = (const float*)d_in[15];
  const float* b2     = (const float*)d_in[16];
  const float* lnf_w  = (const float*)d_in[17];
  const float* lnf_b  = (const float*)d_in[18];
  const float* wc     = (const float*)d_in[19];
  const float* bc     = (const float*)d_in[20];
  float* out = (float*)d_out;

  const size_t BS  = (size_t)NBATCH * SEQ;   // 32768 rows
  const size_t XSZ = BS * D_MODEL;
  float* ws = (float*)d_ws;
  float* x  = ws; ws += XSZ;
  float* n  = ws; ws += XSZ;
  float* q  = ws; ws += XSZ;
  float* k  = ws; ws += XSZ;
  float* v  = ws; ws += XSZ;
  float* sc = ws;                  // scores (B*S*S) and FFN hidden (BS*FF) share
  float* h  = ws;
  float* pooled = ws + (size_t)BS * FFDIM;

  const dim3 blk(256);
  const unsigned rows = (unsigned)BS;
  embed_kernel<<<rows, blk, 0, stream>>>(tokens, emb, pos, x);

  const float inv_scale = 1.0f / sqrtf((float)D_MODEL);
  const long long sQK = (long long)SEQ * D_MODEL;
  const long long sSS = (long long)SEQ * SEQ;

  for (int l = 0; l < NLAYER; ++l) {
    ln_kernel<<<rows, blk, 0, stream>>>(x, ln1_w + l * D_MODEL, ln1_b + l * D_MODEL, n);

    const dim3 gproj(D_MODEL / BN, (unsigned)(BS / BM), 1);
    gemm_wmma_kernel<<<gproj, blk, 0, stream>>>(n, wq + (size_t)l * D_MODEL * D_MODEL,
        bq + l * D_MODEL, nullptr, q, (int)BS, D_MODEL, D_MODEL, 0, 0, 0, 1.f, 0);
    gemm_wmma_kernel<<<gproj, blk, 0, stream>>>(n, wk + (size_t)l * D_MODEL * D_MODEL,
        bk + l * D_MODEL, nullptr, k, (int)BS, D_MODEL, D_MODEL, 0, 0, 0, 1.f, 0);
    gemm_wmma_kernel<<<gproj, blk, 0, stream>>>(n, wvv + (size_t)l * D_MODEL * D_MODEL,
        bv + l * D_MODEL, nullptr, v, (int)BS, D_MODEL, D_MODEL, 0, 0, 0, 1.f, 0);

    // scores = (q @ k^T) / sqrt(D), batched over NBATCH
    const dim3 gsc(SEQ / BN, SEQ / BM, NBATCH);
    gemm_wmma_kernel<<<gsc, blk, 0, stream>>>(q, k, nullptr, nullptr, sc,
        SEQ, SEQ, D_MODEL, sQK, sQK, sSS, inv_scale, 2 /*transB*/);

    softmax_mask_kernel<<<rows, blk, 0, stream>>>(sc, tokens);

    // x += attn @ v, batched
    const dim3 gav(D_MODEL / BN, SEQ / BM, NBATCH);
    gemm_wmma_kernel<<<gav, blk, 0, stream>>>(sc, v, nullptr, x, x,
        SEQ, D_MODEL, SEQ, sSS, sQK, sQK, 1.f, 0);

    ln_kernel<<<rows, blk, 0, stream>>>(x, ln2_w + l * D_MODEL, ln2_b + l * D_MODEL, n);

    const dim3 gff1(FFDIM / BN, (unsigned)(BS / BM), 1);
    gemm_wmma_kernel<<<gff1, blk, 0, stream>>>(n, w1 + (size_t)l * D_MODEL * FFDIM,
        b1 + l * FFDIM, nullptr, h, (int)BS, FFDIM, D_MODEL, 0, 0, 0, 1.f, 1 /*relu*/);

    const dim3 gff2(D_MODEL / BN, (unsigned)(BS / BM), 1);
    gemm_wmma_kernel<<<gff2, blk, 0, stream>>>(h, w2 + (size_t)l * FFDIM * D_MODEL,
        b2 + l * D_MODEL, x, x, (int)BS, D_MODEL, FFDIM, 0, 0, 0, 1.f, 0);
  }

  ln_kernel<<<rows, blk, 0, stream>>>(x, lnf_w, lnf_b, n);
  pool_kernel<<<NBATCH, blk, 0, stream>>>(n, tokens, pooled);
  cls_kernel<<<1, 256, 0, stream>>>(pooled, wc, bc, out);
}
